// WindowAttention_36069135352029
// MI455X (gfx1250) — matrix operand order repaired and checked by reading, not compiled, not run on hardware
//
#include <hip/hip_runtime.h>
#include <hip/hip_bf16.h>

// ---------------- problem constants ----------------
#define DIM_   256
#define WS_    8
#define H_     8
#define HD_    32
#define NREF_  30
#define RB_    4
#define NWIN_  144
#define N_     64            // WS*WS
#define P_     9216          // NWIN*N
#define MTOK_  36864         // B_*N = RB*P
#define PLANE_ 276480        // P_*NREF_
#define SCALE_ 0.17677669529663687f

typedef __attribute__((ext_vector_type(16))) _Float16 v16h;
typedef __attribute__((ext_vector_type(8)))  float    v8f;

// ---------------- workspace layout (bytes) ----------------
// half buffers are 2B/elem, float 4B/elem
#define OFF_XH       ((size_t)0)                       // 36864*256 f16 (later reused as qnew16)
#define OFF_QKVW     ((size_t)18874368)                // 768*256 f16
#define OFF_PROJW    ((size_t)19267584)                // 256*256 f16
#define OFF_REFQ     ((size_t)19398656)                // 4*8*32*32 f16 (K-padded)
#define OFF_REFVT    ((size_t)19464192)                // 4*8*32*32 f16 (transposed, K-padded)
#define OFF_RELBIAS  ((size_t)19529728)                // 8*64*64 f32
#define OFF_STATS    ((size_t)19660800)                // 64 f32
#define OFF_Q16      ((size_t)19661056)                // 32*9216*32 f16 (later reused as out16)
#define OFF_K16      ((size_t)38535424)                // 4608*64*32 f16
#define OFF_VT16     ((size_t)57409792)                // 4608*32*64 f16
#define OFF_REFATTN  ((size_t)76284160)                // 32*9216*30 f32
#define OFF_UPD      ((size_t)111673600)               // 32*9216*30 f32 (later reused as softA f16)
#define OFF_QNEW     OFF_XH
#define OFF_OUT16    OFF_Q16
#define OFF_SOFTA    OFF_UPD
// total required: ~147.1 MB

#define DEV static __device__ __forceinline__

// ---------------- WMMA fragment helpers ----------------
// A: 16x32 f16 row-major [m][k], elements e are K = 16*(lane/16) + e (consecutive)
DEV v16h load_a_frag(const _Float16* __restrict__ A, int lda, int lane) {
  const int m = lane & 15, g = lane >> 4;
  v16h a;
#pragma unroll
  for (int r = 0; r < 8; ++r) {
    const int k = (g << 4) + (r << 1);
    a[2 * r]     = A[m * lda + k];
    a[2 * r + 1] = A[m * lda + k + 1];
  }
  return a;
}

// B: 32x16 f16, supplied as B^T row-major [n][k]; lane holds column n = lane%16,
// elements e are K = 16*(lane/16) + e (consecutive)
DEV v16h load_bt_frag(const _Float16* __restrict__ Bt, int ldb, int lane) {
  const int n = lane & 15, g = lane >> 4;
  v16h b;
#pragma unroll
  for (int e = 0; e < 16; ++e) b[e] = Bt[n * ldb + (g << 4) + e];
  return b;
}

DEV v8f wmma_f16(v16h a, v16h b, v8f c) {
  return __builtin_amdgcn_wmma_f32_16x16x32_f16(false, a, false, b, (short)0, c,
                                                false, false);
}

// ---------------- small prep kernels ----------------
__global__ void k_prep(const float* __restrict__ qkv_w, const float* __restrict__ proj_w,
                       const float* __restrict__ bias_table, const int* __restrict__ rel_index,
                       _Float16* __restrict__ qkv_wh, _Float16* __restrict__ proj_wh,
                       float* __restrict__ relbias) {
  int id = blockIdx.x * blockDim.x + threadIdx.x;
  if (id < 196608) { qkv_wh[id] = (_Float16)qkv_w[id]; return; }
  id -= 196608;
  if (id < 65536) { proj_wh[id] = (_Float16)proj_w[id]; return; }
  id -= 65536;
  if (id < 32768) {                       // relbias[h][i][j] = bias_table[rel_index[i][j]][h]
    const int h = id >> 12, ij = id & 4095;
    relbias[id] = bias_table[rel_index[ij] * H_ + h];
  }
}

__global__ void k_f2h(const float* __restrict__ x, _Float16* __restrict__ xh, int n) {
  int i = blockIdx.x * blockDim.x + threadIdx.x;
  const int stride = gridDim.x * blockDim.x;
  for (; i < n; i += stride) xh[i] = (_Float16)x[i];
}

// ref = x_ref @ ref_qk_w.T + b ; ref_q = mu + exp(ls)*ref[:, :30] ; ref_v = ref[:, 30:]
// writes K/N-padded f16 operand tensors (rows 30,31 zero)
__global__ void k_refprep(const float* __restrict__ x_ref, const float* __restrict__ ref_qk_w,
                          const float* __restrict__ ref_qk_b, const float* __restrict__ diff_mu,
                          const float* __restrict__ diff_ls,
                          _Float16* __restrict__ refq16, _Float16* __restrict__ refvT16) {
  const int id = blockIdx.x * blockDim.x + threadIdx.x;   // 65536 total
  if (id >= 65536) return;
  const int sel = id >> 15;          // 0 = ref_q, 1 = ref_v
  const int rem = id & 32767;
  const int b   = rem >> 13;         // /8192
  const int r32 = (rem >> 8) & 31;
  const int dim = rem & 255;
  float val = 0.f;
  if (r32 < NREF_) {
    const int j = (sel == 0) ? r32 : NREF_ + r32;
    float acc = ref_qk_b[j];
    const float* xr = x_ref + (b * DIM_ + dim) * NREF_;
    for (int t = 0; t < NREF_; ++t) acc += xr[t] * ref_qk_w[j * NREF_ + t];
    val = (sel == 0) ? (diff_mu[dim] + expf(diff_ls[dim]) * acc) : acc;
  }
  const int hh = dim >> 5, d = dim & 31;
  if (sel == 0) refq16 [((b * H_ + hh) * 32 + r32) * 32 + d] = (_Float16)val; // [b][h][r][d]
  else          refvT16[((b * H_ + hh) * 32 + d) * 32 + r32] = (_Float16)val; // [b][h][d][r]
}

// ---------------- qkv projection GEMM: (36864x256)x(256x768) ----------------
__global__ __launch_bounds__(128)
void k_gemm_qkv(const _Float16* __restrict__ xh, const _Float16* __restrict__ wh,
                const float* __restrict__ qkv_b,
                _Float16* __restrict__ q16, _Float16* __restrict__ k16,
                _Float16* __restrict__ vT16) {
  const int lane = threadIdx.x & 31, wv = threadIdx.x >> 5;
  const int m0 = blockIdx.x * 64 + wv * 16;
  const int n0 = blockIdx.y * 16;
  v8f c = {};
  const _Float16* Ap = xh + (size_t)m0 * DIM_;
  const _Float16* Bp = wh + (size_t)n0 * DIM_;
#pragma unroll
  for (int k0 = 0; k0 < DIM_; k0 += 32) {
    v16h a = load_a_frag(Ap + k0, DIM_, lane);
    v16h b = load_bt_frag(Bp + k0, DIM_, lane);
    c = wmma_f16(a, b, c);
  }
  const int ncol  = n0 + (lane & 15);
  const int which = ncol >> 8;            // 0=q 1=k 2=v
  const int hh    = (ncol >> 5) & 7;
  const int d     = ncol & 31;
  const float bias = qkv_b[ncol];
  const int g = lane >> 4;
#pragma unroll
  for (int r = 0; r < 8; ++r) {
    const int gtok = m0 + r + 8 * g;
    const float v  = c[r] + bias;
    const int b_ = gtok / P_;
    const int p  = gtok - b_ * P_;
    const int bw = gtok >> 6;
    const int nn = gtok & 63;
    if (which == 0)      q16 [((size_t)(b_ * H_ + hh) * P_ + p) * HD_ + d]  = (_Float16)v;
    else if (which == 1) k16 [((size_t)(bw * H_ + hh) * N_ + nn) * HD_ + d] = (_Float16)v;
    else                 vT16[((size_t)(bw * H_ + hh) * HD_ + d) * N_ + nn] = (_Float16)v;
  }
}

// ---------------- ref_attn = q @ ref_q^T, per (b,h): (9216x32)x(32x32) ----------------
__global__ __launch_bounds__(128)
void k_gemm_ra(const _Float16* __restrict__ q16, const _Float16* __restrict__ refq16,
               float* __restrict__ ref_attn) {
  const int lane = threadIdx.x & 31, wv = threadIdx.x >> 5;
  const int bh = blockIdx.z;
  const int m0 = blockIdx.x * 64 + wv * 16;
  const int n0 = blockIdx.y * 16;
  v16h a = load_a_frag(q16 + ((size_t)bh * P_ + m0) * HD_, HD_, lane);
  v16h b = load_bt_frag(refq16 + (size_t)(bh * 32 + n0) * 32, 32, lane);
  v8f c = {};
  c = wmma_f16(a, b, c);
  const int rcol = n0 + (lane & 15);
  if (rcol < NREF_) {
    const int g = lane >> 4;
#pragma unroll
    for (int r = 0; r < 8; ++r)
      ref_attn[((size_t)bh * P_ + m0 + r + 8 * g) * NREF_ + rcol] = c[r];
  }
}

// ---------------- conv diffusion refinement ----------------
__global__ void k_zero_stats(float* __restrict__ stats) {
  if (threadIdx.x < 64) stats[threadIdx.x] = 0.f;
}

__global__ __launch_bounds__(256)
void k_conv(const float* __restrict__ ref_attn, const float* __restrict__ conv_w,
            const float* __restrict__ conv_b, float* __restrict__ upd,
            float* __restrict__ stats) {
  const int bh = blockIdx.y;
  const int h = bh & 7, b = bh >> 3;
  float s = 0.f, ss = 0.f;
  const int base = blockIdx.x * 1024;
  for (int i = 0; i < 4; ++i) {
    const int e = base + i * 256 + threadIdx.x;
    const int y = e / NREF_, x = e - y * NREF_;
    float acc = conv_b[h];
    for (int ci = 0; ci < H_; ++ci) {
      const float* src  = ref_attn + (size_t)(b * H_ + ci) * PLANE_;
      const float* wrow = conv_w + (h * H_ + ci) * 9;
#pragma unroll
      for (int ky = 0; ky < 3; ++ky) {
        const int yy = y + ky - 1;
        if ((unsigned)yy < (unsigned)P_) {
#pragma unroll
          for (int kx = 0; kx < 3; ++kx) {
            const int xx = x + kx - 1;
            if ((unsigned)xx < (unsigned)NREF_) acc += src[yy * NREF_ + xx] * wrow[ky * 3 + kx];
          }
        }
      }
    }
    upd[(size_t)bh * PLANE_ + e] = acc;
    s += acc; ss += acc * acc;
  }
  __shared__ float s1[256], s2[256];
  s1[threadIdx.x] = s; s2[threadIdx.x] = ss;
  __syncthreads();
  for (int off = 128; off > 0; off >>= 1) {
    if (threadIdx.x < off) { s1[threadIdx.x] += s1[threadIdx.x + off];
                             s2[threadIdx.x] += s2[threadIdx.x + off]; }
    __syncthreads();
  }
  if (threadIdx.x == 0) { atomicAdd(&stats[2 * bh], s1[0]); atomicAdd(&stats[2 * bh + 1], s2[0]); }
}

__global__ __launch_bounds__(256)
void k_apply(const float* __restrict__ upd, const float* __restrict__ stats,
             float* __restrict__ ref_attn) {
  const int bh = blockIdx.y;
  const float inv  = 1.f / (float)PLANE_;
  const float mean = stats[2 * bh] * inv;
  const float var  = fmaxf(stats[2 * bh + 1] * inv - mean * mean, 0.f);
  const float rstd = rsqrtf(var + 1e-5f);
  const int base = blockIdx.x * 1024 + threadIdx.x;
  for (int i = 0; i < 4; ++i) {
    const size_t e = (size_t)bh * PLANE_ + base + i * 256;
    const float u = (upd[e] - mean) * rstd;
    ref_attn[e] += 0.5f * u * (1.f + erff(u * 0.70710678f));   // exact gelu
  }
}

// ---------------- softmax over NREF -> f16 A operand (K padded to 32) ----------------
__global__ void k_softmax_ref(const float* __restrict__ ref_attn, _Float16* __restrict__ softA) {
  const int row = blockIdx.x * blockDim.x + threadIdx.x;   // 294912 rows
  if (row >= 32 * P_) return;
  const float* src = ref_attn + (size_t)row * NREF_;
  float v[NREF_], m = -1e30f;
  for (int r = 0; r < NREF_; ++r) { v[r] = src[r]; m = fmaxf(m, v[r]); }
  float s = 0.f;
  for (int r = 0; r < NREF_; ++r) { v[r] = expf(v[r] - m); s += v[r]; }
  const float inv = 1.f / s;
  _Float16* dst = softA + (size_t)row * 32;
  for (int r = 0; r < NREF_; ++r) dst[r] = (_Float16)(v[r] * inv);
  dst[30] = (_Float16)0.f; dst[31] = (_Float16)0.f;
}

// ---------------- q_new = softmax(ref_attn) @ ref_v * SCALE ----------------
__global__ __launch_bounds__(128)
void k_gemm_qnew(const _Float16* __restrict__ softA, const _Float16* __restrict__ refvT16,
                 _Float16* __restrict__ qnew16) {
  const int lane = threadIdx.x & 31, wv = threadIdx.x >> 5;
  const int bh = blockIdx.z;
  const int m0 = blockIdx.x * 64 + wv * 16;
  const int n0 = blockIdx.y * 16;
  v16h a = load_a_frag(softA + ((size_t)bh * P_ + m0) * 32, 32, lane);
  v16h b = load_bt_frag(refvT16 + (size_t)(bh * 32 + n0) * 32, 32, lane);
  v8f c = {};
  c = wmma_f16(a, b, c);
  const int n = n0 + (lane & 15), g = lane >> 4;
#pragma unroll
  for (int r = 0; r < 8; ++r)
    qnew16[((size_t)bh * P_ + m0 + r + 8 * g) * HD_ + n] = (_Float16)(c[r] * SCALE_);
}

// ---------------- fused window attention: scores + bias + softmax + P@V ----------------
__global__ __launch_bounds__(128)
void k_attn(const _Float16* __restrict__ qnew16, const _Float16* __restrict__ k16,
            const _Float16* __restrict__ vT16, const float* __restrict__ relbias,
            const float* __restrict__ mask, _Float16* __restrict__ out16) {
  const int lane = threadIdx.x & 31, wv = threadIdx.x >> 5;
  const int bwh = blockIdx.x;
  const int bw = bwh >> 3, h = bwh & 7;
  const int b = bw / NWIN_, w = bw - b * NWIN_;
  const int m0 = wv * 16;
  const int g = lane >> 4, nlo = lane & 15;
  __shared__ _Float16 P[64 * 64];

  v16h a = load_a_frag(qnew16 + (((size_t)(b * H_ + h) * P_) + w * 64 + m0) * HD_, HD_, lane);
  float sreg[4][8];
#pragma unroll
  for (int t = 0; t < 4; ++t) {
    v16h bf = load_bt_frag(k16 + (size_t)((bw * H_ + h) * N_ + t * 16) * HD_, HD_, lane);
    v8f c = {};
    c = wmma_f16(a, bf, c);
    const int n = t * 16 + nlo;
#pragma unroll
    for (int r = 0; r < 8; ++r) {
      const int m = m0 + r + 8 * g;
      sreg[t][r] = c[r] + relbias[(h * 64 + m) * 64 + n] + mask[(w * 64 + m) * 64 + n];
    }
  }
  // per-row softmax: row m lives in one 16-lane group; xor masks < 16 stay in-group
#pragma unroll
  for (int r = 0; r < 8; ++r) {
    float mx = fmaxf(fmaxf(sreg[0][r], sreg[1][r]), fmaxf(sreg[2][r], sreg[3][r]));
    for (int off = 1; off < 16; off <<= 1) mx = fmaxf(mx, __shfl_xor(mx, off, 32));
    float sum = 0.f;
#pragma unroll
    for (int t = 0; t < 4; ++t) { sreg[t][r] = expf(sreg[t][r] - mx); sum += sreg[t][r]; }
    for (int off = 1; off < 16; off <<= 1) sum += __shfl_xor(sum, off, 32);
    const float inv = 1.f / sum;
    const int m = m0 + r + 8 * g;
#pragma unroll
    for (int t = 0; t < 4; ++t) P[m * 64 + t * 16 + nlo] = (_Float16)(sreg[t][r] * inv);
  }
  __syncthreads();
#pragma unroll
  for (int dt = 0; dt < 2; ++dt) {
    v8f c = {};
#pragma unroll
    for (int ks = 0; ks < 64; ks += 32) {
      v16h pa = load_a_frag(&P[m0 * 64 + ks], 64, lane);
      v16h vb = load_bt_frag(vT16 + (size_t)((bw * H_ + h) * HD_ + dt * 16) * N_ + ks, N_, lane);
      c = wmma_f16(pa, vb, c);
    }
    const int col = h * HD_ + dt * 16 + nlo;
#pragma unroll
    for (int r = 0; r < 8; ++r) {
      const int gtok = bw * 64 + m0 + r + 8 * g;
      out16[(size_t)gtok * DIM_ + col] = (_Float16)c[r];
    }
  }
}

// ---------------- output projection: (36864x256)x(256x256) + bias ----------------
__global__ __launch_bounds__(128)
void k_gemm_proj(const _Float16* __restrict__ out16, const _Float16* __restrict__ pwh,
                 const float* __restrict__ proj_b, float* __restrict__ out) {
  const int lane = threadIdx.x & 31, wv = threadIdx.x >> 5;
  const int m0 = blockIdx.x * 64 + wv * 16;
  const int n0 = blockIdx.y * 16;
  v8f c = {};
#pragma unroll
  for (int k0 = 0; k0 < DIM_; k0 += 32) {
    v16h a = load_a_frag(out16 + (size_t)m0 * DIM_ + k0, DIM_, lane);
    v16h b = load_bt_frag(pwh + (size_t)n0 * DIM_ + k0, DIM_, lane);
    c = wmma_f16(a, b, c);
  }
  const int n = n0 + (lane & 15), g = lane >> 4;
  const float bias = proj_b[n];
#pragma unroll
  for (int r = 0; r < 8; ++r)
    out[(size_t)(m0 + r + 8 * g) * DIM_ + n] = c[r] + bias;
}

// ---------------- launcher ----------------
extern "C" void kernel_launch(void* const* d_in, const int* in_sizes, int n_in,
                              void* d_out, int out_size, void* d_ws, size_t ws_size,
                              hipStream_t stream) {
  (void)in_sizes; (void)n_in; (void)out_size; (void)ws_size;
  const float* x          = (const float*)d_in[0];
  const float* mask       = (const float*)d_in[1];
  const float* x_ref      = (const float*)d_in[2];
  const float* qkv_w      = (const float*)d_in[3];
  const float* qkv_b      = (const float*)d_in[4];
  const float* proj_w     = (const float*)d_in[5];
  const float* proj_b     = (const float*)d_in[6];
  const float* bias_table = (const float*)d_in[7];
  const float* diff_mu    = (const float*)d_in[8];
  const float* diff_ls    = (const float*)d_in[9];
  const float* ref_qk_w   = (const float*)d_in[10];
  const float* ref_qk_b   = (const float*)d_in[11];
  const float* conv_w     = (const float*)d_in[12];
  const float* conv_b     = (const float*)d_in[13];
  const int*   rel_index  = (const int*)d_in[14];
  float* out = (float*)d_out;
  char*  ws  = (char*)d_ws;

  _Float16* xh      = (_Float16*)(ws + OFF_XH);
  _Float16* qkvwh   = (_Float16*)(ws + OFF_QKVW);
  _Float16* projwh  = (_Float16*)(ws + OFF_PROJW);
  _Float16* refq16  = (_Float16*)(ws + OFF_REFQ);
  _Float16* refvT16 = (_Float16*)(ws + OFF_REFVT);
  float*    relbias = (float*)(ws + OFF_RELBIAS);
  float*    stats   = (float*)(ws + OFF_STATS);
  _Float16* q16     = (_Float16*)(ws + OFF_Q16);
  _Float16* k16     = (_Float16*)(ws + OFF_K16);
  _Float16* vT16    = (_Float16*)(ws + OFF_VT16);
  float*    refattn = (float*)(ws + OFF_REFATTN);
  float*    upd     = (float*)(ws + OFF_UPD);
  _Float16* softA   = (_Float16*)(ws + OFF_SOFTA);   // reuses upd region
  _Float16* qnew16  = (_Float16*)(ws + OFF_QNEW);    // reuses xh region
  _Float16* out16   = (_Float16*)(ws + OFF_OUT16);   // reuses q16 region

  k_prep<<<1152, 256, 0, stream>>>(qkv_w, proj_w, bias_table, rel_index, qkvwh, projwh, relbias);
  k_f2h<<<9216, 256, 0, stream>>>(x, xh, MTOK_ * DIM_);
  k_refprep<<<256, 256, 0, stream>>>(x_ref, ref_qk_w, ref_qk_b, diff_mu, diff_ls,
                                     refq16, refvT16);

  dim3 gq(MTOK_ / 64, 768 / 16);
  k_gemm_qkv<<<gq, 128, 0, stream>>>(xh, qkvwh, qkv_b, q16, k16, vT16);

  dim3 gra(P_ / 64, 2, RB_ * H_);
  k_gemm_ra<<<gra, 128, 0, stream>>>(q16, refq16, refattn);

  dim3 gc(PLANE_ / 1024, RB_ * H_);
  for (int it = 0; it < 3; ++it) {
    k_zero_stats<<<1, 64, 0, stream>>>(stats);
    k_conv<<<gc, 256, 0, stream>>>(refattn, conv_w, conv_b, upd, stats);
    k_apply<<<gc, 256, 0, stream>>>(upd, stats, refattn);
  }

  k_softmax_ref<<<1152, 256, 0, stream>>>(refattn, softA);
  k_gemm_qnew<<<gra, 128, 0, stream>>>(softA, refvT16, qnew16);
  k_attn<<<RB_ * NWIN_ * H_, 128, 0, stream>>>(qnew16, k16, vT16, relbias, mask, out16);

  dim3 gp(MTOK_ / 64, DIM_ / 16);
  k_gemm_proj<<<gp, 128, 0, stream>>>(out16, projwh, proj_b, out);
}
